// MultiHeadAttention_8624294330377
// MI455X (gfx1250) — compile-verified, matrix operand order checked by reading, not run on hardware
//
#include <hip/hip_runtime.h>
#include <cstdint>
#include <cstddef>

// Problem constants (from reference)
#define NH   16
#define DKV  64
#define DMOD 1024
#define NB   4
#define NS   4096
#define MTOT (NB * NS)   // 16384

typedef __attribute__((ext_vector_type(16))) __bf16 v16bf;
typedef __attribute__((ext_vector_type(8)))  float  v8f;

union Frag {
    uint4 u[2];
    v16bf v;
};

__device__ __forceinline__ unsigned short f2bf(float x) {
    unsigned u = __float_as_uint(x);
    unsigned r = u + 0x7FFFu + ((u >> 16) & 1u);   // round-to-nearest-even
    return (unsigned short)(r >> 16);
}
__device__ __forceinline__ unsigned pack2bf(float a, float b) {
    return (unsigned)f2bf(a) | ((unsigned)f2bf(b) << 16);
}
__device__ __forceinline__ float bflo(unsigned u) { return __uint_as_float(u << 16); }
__device__ __forceinline__ float bfhi(unsigned u) { return __uint_as_float(u & 0xFFFF0000u); }

// gfx1250 async global->LDS copy, 16 bytes per lane, GVS addressing:
//   LDS[vdst + 0] = MEM[saddr + voffset + 0 .. 15]
// Tracked by ASYNCcnt (s_wait_asynccnt).
__device__ __forceinline__ void async_copy16(unsigned lds_byte_off,
                                             const void* sbase, unsigned voff) {
    asm volatile("global_load_async_to_lds_b128 %0, %1, %2 offset:0"
                 :: "v"(lds_byte_off), "v"(voff), "s"(sbase) : "memory");
}
__device__ __forceinline__ void wait_async0() {
    asm volatile("s_wait_asynccnt 0x0" ::: "memory");
}

// ---------------------------------------------------------------------------
// Kernel 0: x [B,S,D] f32 -> bf16 (one-time; amortized over 16 head-blocks)
// ---------------------------------------------------------------------------
__global__ __launch_bounds__(256) void pack_x(const float* __restrict__ x,
                                              uint4* __restrict__ xb) {
    const size_t i = (size_t)blockIdx.x * 256 + threadIdx.x;  // 8 floats each
    const float4* src = (const float4*)x + i * 2;
    float4 f0 = src[0], f1 = src[1];
    uint4 p;
    p.x = pack2bf(f0.x, f0.y); p.y = pack2bf(f0.z, f0.w);
    p.z = pack2bf(f1.x, f1.y); p.w = pack2bf(f1.z, f1.w);
    xb[i] = p;
}

// ---------------------------------------------------------------------------
// Kernel 1: pack Wk|Wq|Wv ([H,D,64] f32 each) -> Wp [H,192,1024] bf16 (N-major)
//           and biasP [H,192] f32
// ---------------------------------------------------------------------------
__global__ __launch_bounds__(256) void pack_qkv_w(
    const float* __restrict__ Wk, const float* __restrict__ Wq,
    const float* __restrict__ Wv, const float* __restrict__ bk,
    const float* __restrict__ bq, const float* __restrict__ bv,
    unsigned short* __restrict__ Wp, float* __restrict__ biasP) {
    const int h = blockIdx.x;
    const int t = threadIdx.x;
    const float* srcs[3] = {Wk, Wq, Wv};
    for (int g = 0; g < 3; ++g) {
        const float* src = srcs[g] + (size_t)h * DMOD * DKV;
        unsigned short* dst = Wp + ((size_t)h * 192 + g * 64) * DMOD;
        for (int idx = t; idx < DMOD * DKV; idx += 256) {
            int d = idx >> 6;
            int n = idx & 63;
            dst[(size_t)n * DMOD + d] = f2bf(src[idx]);
        }
    }
    if (t < 192) {
        float bvv;
        if (t < 64)       bvv = bk[h * 64 + t];
        else if (t < 128) bvv = bq[h * 64 + t - 64];
        else              bvv = bv[h * 64 + t - 128];
        biasP[h * 192 + t] = bvv;
    }
}

// ---------------------------------------------------------------------------
// Kernel 2: transpose Wo [1024(K) x 1024(N)] f32 -> WoT [N][K] bf16
// ---------------------------------------------------------------------------
__global__ __launch_bounds__(256) void pack_wo(
    const float* __restrict__ Wo, unsigned short* __restrict__ WoT) {
    const int n = blockIdx.x;
    const int t = threadIdx.x;
    for (int i = 0; i < 4; ++i) {
        int k = t + 256 * i;
        WoT[(size_t)n * 1024 + k] = f2bf(Wo[(size_t)k * 1024 + n]);
    }
}

// ---------------------------------------------------------------------------
// Kernel 3: fused QKV projection GEMM + bias + score + V emission.
// Grid: (128 row-blocks, 16 heads). Block: 256 threads (8 waves, 2x4 grid).
// Tile: M=128, N=192 (K|Q|V), K-step 32; async double-buffered staging.
// LDS: bufA0[0,8K) bufB0[8K,20K) bufA1[20K,28K) bufB1[28K,40K); C overlay 96K.
// ---------------------------------------------------------------------------
__global__ __launch_bounds__(256) void qkv_score_v(
    const unsigned short* __restrict__ xb, const unsigned short* __restrict__ Wp,
    const float* __restrict__ biasP, float* __restrict__ scores,
    unsigned* __restrict__ VxP /* bf16 pairs, [B,H,S,32] uints */) {
    extern __shared__ char smem[];
    float* Cs = (float*)smem;   // epilogue overlay: 128 x 192 f32

    const int t    = threadIdx.x;
    const int lane = t & 31;
    const int wid  = t >> 5;
    const int h     = blockIdx.y;
    const int mBase = blockIdx.x * 128;

    const int m0w = (wid >> 2) * 64;   // wave M origin
    const int n0w = (wid & 3) * 48;    // wave N origin

    v8f acc[12];
    for (int i = 0; i < 12; ++i)
        for (int j = 0; j < 8; ++j) acc[i][j] = 0.0f;

    // byte bases for async staging (row stride 2048 B = 1024 bf16)
    const char* xbase = (const char*)xb + (size_t)mBase * 2048;
    const char* wbase = (const char*)Wp + (size_t)h * 192 * 2048;

    auto issue_tiles = [&](int ks, unsigned aB, unsigned bB) {
        const int ksb = ks * 2;
        // A tile: 128 rows x 64B = 512 x 16B chunks, 2 per thread
        for (int i = 0; i < 2; ++i) {
            int c = t + 256 * i;
            unsigned vo = (unsigned)((c >> 2) * 2048 + (c & 3) * 16 + ksb);
            async_copy16(aB + (unsigned)((c >> 2) * 64 + (c & 3) * 16), xbase, vo);
        }
        // B tile: 192 rows x 64B = 768 x 16B chunks, 3 per thread
        for (int c = t; c < 768; c += 256) {
            unsigned vo = (unsigned)((c >> 2) * 2048 + (c & 3) * 16 + ksb);
            async_copy16(bB + (unsigned)((c >> 2) * 64 + (c & 3) * 16), wbase, vo);
        }
    };

    issue_tiles(0, 0u, 8192u);

    for (int it = 0; it < 32; ++it) {
        wait_async0();
        __syncthreads();
        if (it + 1 < 32) {
            unsigned nb = (it & 1) ? 0u : 20480u;        // next buffers
            issue_tiles((it + 1) * 32, nb, nb + 8192u);
        }
        const unsigned aB = (it & 1) ? 20480u : 0u;      // current buffers
        const unsigned bB = aB + 8192u;

        const int lr = lane & 15;
        const int hi = lane >> 4;
        Frag a[4], b[3];
        for (int mt = 0; mt < 4; ++mt) {
            const char* p = smem + aB + (m0w + mt * 16 + lr) * 64 + hi * 16;
            a[mt].u[0] = *(const uint4*)(p);
            a[mt].u[1] = *(const uint4*)(p + 32);
        }
        for (int nt = 0; nt < 3; ++nt) {
            const char* p = smem + bB + (n0w + nt * 16 + lr) * 64 + hi * 16;
            b[nt].u[0] = *(const uint4*)(p);
            b[nt].u[1] = *(const uint4*)(p + 32);
        }
        for (int mt = 0; mt < 4; ++mt)
            for (int nt = 0; nt < 3; ++nt)
                acc[mt * 3 + nt] = __builtin_amdgcn_wmma_f32_16x16x32_bf16(
                    false, a[mt].v, false, b[nt].v, (short)0, acc[mt * 3 + nt],
                    false, false);
    }

    __syncthreads();
    // --- dump C (+bias) to LDS: [128][192] f32 ---
    {
        const int lr = lane & 15;
        const int mo = (lane >> 4) * 8;
        for (int nt = 0; nt < 3; ++nt) {
            int col = n0w + nt * 16 + lr;
            float bcol = biasP[h * 192 + col];
            for (int mt = 0; mt < 4; ++mt) {
                int rbase = m0w + mt * 16 + mo;
                v8f c = acc[mt * 3 + nt];
                for (int r = 0; r < 8; ++r)
                    Cs[(rbase + r) * 192 + col] = c[r] + bcol;
            }
        }
    }
    __syncthreads();
    // --- scores = sum(K*Q)/8 ; emit V rows as bf16 ---
    {
        const int row  = t >> 1;
        const int half = t & 1;
        const float* crow = Cs + row * 192;
        float s = 0.0f;
        for (int j = 0; j < 32; ++j)
            s += crow[half * 32 + j] * crow[64 + half * 32 + j];
        s += __shfl_xor(s, 1, 32);
        const int    mg  = mBase + row;
        const int    bI  = mg >> 12;
        const int    sI  = mg & 4095;
        const size_t bhs = ((size_t)bI * NH + h) * NS + sI;
        if (half == 0) scores[bhs] = s * 0.125f;   // 1/sqrt(64)
        unsigned* vdst = VxP + bhs * 32 + half * 16;
        const float* vsrc = crow + 128 + half * 32;
        for (int j = 0; j < 16; ++j)
            vdst[j] = pack2bf(vsrc[2 * j], vsrc[2 * j + 1]);
    }
}

// ---------------------------------------------------------------------------
// Kernel 4: softmax over S per (b,h), then scale V rows -> concat [B,S,1024] bf16
// ---------------------------------------------------------------------------
__global__ __launch_bounds__(256) void softmax_scale(
    const float* __restrict__ scores, const unsigned* __restrict__ VxP,
    unsigned* __restrict__ concatP) {
    __shared__ float red[8];
    const int t = threadIdx.x;
    const int lane = t & 31, wid = t >> 5;
    const int bI = blockIdx.x >> 4;
    const int h  = blockIdx.x & 15;
    const size_t base = ((size_t)bI * NH + h) * NS;

    float v[16];
    float m = -3.4e38f;
    for (int i = 0; i < 16; ++i) {
        v[i] = scores[base + t + 256 * i];
        m = fmaxf(m, v[i]);
    }
    for (int o = 16; o; o >>= 1) m = fmaxf(m, __shfl_xor(m, o, 32));
    if (lane == 0) red[wid] = m;
    __syncthreads();
    if (t == 0) {
        float mm = red[0];
        for (int i = 1; i < 8; ++i) mm = fmaxf(mm, red[i]);
        red[0] = mm;
    }
    __syncthreads();
    m = red[0];
    float sum = 0.0f;
    for (int i = 0; i < 16; ++i) { v[i] = __expf(v[i] - m); sum += v[i]; }
    for (int o = 16; o; o >>= 1) sum += __shfl_xor(sum, o, 32);
    __syncthreads();
    if (lane == 0) red[wid] = sum;
    __syncthreads();
    if (t == 0) {
        float ss = red[0];
        for (int i = 1; i < 8; ++i) ss += red[i];
        red[0] = ss;
    }
    __syncthreads();
    const float inv = 1.0f / red[0];

    for (int i = 0; i < 16; ++i) {
        const int   row = t + 256 * i;
        const float w   = v[i] * inv;
        const unsigned* src = VxP + (base + row) * 32;
        unsigned*       dst = concatP + ((size_t)bI * NS + row) * 512 + h * 32;
        for (int j = 0; j < 32; ++j) {
            unsigned u = src[j];
            dst[j] = pack2bf(bflo(u) * w, bfhi(u) * w);
        }
    }
}

// ---------------------------------------------------------------------------
// Kernel 5: output GEMM: concat[16384,1024] bf16 x WoT(N-major) + bo -> out f32
// Async double-buffered staging. LDS: A0[0,8K) B0[8K,16K) A1[16K,24K) B1[24K,32K)
// ---------------------------------------------------------------------------
__global__ __launch_bounds__(256) void out_gemm(
    const unsigned short* __restrict__ A, const unsigned short* __restrict__ Bt,
    const float* __restrict__ bo, float* __restrict__ out) {
    extern __shared__ char smem[];
    const int t = threadIdx.x;
    const int lane = t & 31, wid = t >> 5;
    const int mBase = blockIdx.x * 128;
    const int nBase = blockIdx.y * 128;
    const int m0w = (wid >> 2) * 64;
    const int n0w = (wid & 3) * 32;

    v8f acc[8];
    for (int i = 0; i < 8; ++i)
        for (int j = 0; j < 8; ++j) acc[i][j] = 0.0f;

    const char* abase = (const char*)A  + (size_t)mBase * 2048;
    const char* bbase = (const char*)Bt + (size_t)nBase * 2048;

    auto issue_tiles = [&](int ks, unsigned aB, unsigned bB) {
        const int ksb = ks * 2;
        for (int i = 0; i < 2; ++i) {
            int c = t + 256 * i;
            unsigned vo = (unsigned)((c >> 2) * 2048 + (c & 3) * 16 + ksb);
            unsigned lo = (unsigned)((c >> 2) * 64 + (c & 3) * 16);
            async_copy16(aB + lo, abase, vo);
            async_copy16(bB + lo, bbase, vo);
        }
    };

    issue_tiles(0, 0u, 8192u);

    for (int it = 0; it < 32; ++it) {
        wait_async0();
        __syncthreads();
        if (it + 1 < 32) {
            unsigned nb = (it & 1) ? 0u : 16384u;
            issue_tiles((it + 1) * 32, nb, nb + 8192u);
        }
        const unsigned aB = (it & 1) ? 16384u : 0u;
        const unsigned bB = aB + 8192u;

        const int lr = lane & 15, hi = lane >> 4;
        Frag a[4], b[2];
        for (int mt = 0; mt < 4; ++mt) {
            const char* p = smem + aB + (m0w + mt * 16 + lr) * 64 + hi * 16;
            a[mt].u[0] = *(const uint4*)p;
            a[mt].u[1] = *(const uint4*)(p + 32);
        }
        for (int nt = 0; nt < 2; ++nt) {
            const char* p = smem + bB + (n0w + nt * 16 + lr) * 64 + hi * 16;
            b[nt].u[0] = *(const uint4*)p;
            b[nt].u[1] = *(const uint4*)(p + 32);
        }
        for (int mt = 0; mt < 4; ++mt)
            for (int nt = 0; nt < 2; ++nt)
                acc[mt * 2 + nt] = __builtin_amdgcn_wmma_f32_16x16x32_bf16(
                    false, a[mt].v, false, b[nt].v, (short)0, acc[mt * 2 + nt],
                    false, false);
    }

    const int lr = lane & 15;
    const int mo = (lane >> 4) * 8;
    for (int nt = 0; nt < 2; ++nt) {
        int col = nBase + n0w + nt * 16 + lr;
        float bc = bo[col];
        for (int mt = 0; mt < 4; ++mt) {
            int rbase = mBase + m0w + mt * 16 + mo;
            v8f c = acc[mt * 2 + nt];
            for (int r = 0; r < 8; ++r)
                out[(size_t)(rbase + r) * 1024 + col] = c[r] + bc;
        }
    }
}

// ---------------------------------------------------------------------------
// Host launcher
// ---------------------------------------------------------------------------
extern "C" void kernel_launch(void* const* d_in, const int* in_sizes, int n_in,
                              void* d_out, int out_size, void* d_ws, size_t ws_size,
                              hipStream_t stream) {
    const float* x  = (const float*)d_in[0];
    const float* Wk = (const float*)d_in[1];
    const float* bk = (const float*)d_in[2];
    const float* Wq = (const float*)d_in[3];
    const float* bq = (const float*)d_in[4];
    const float* Wv = (const float*)d_in[5];
    const float* bv = (const float*)d_in[6];
    const float* Wo = (const float*)d_in[7];
    const float* bo = (const float*)d_in[8];
    float* out = (float*)d_out;

    char* ws = (char*)d_ws;
    unsigned short* Wp      = (unsigned short*)(ws);             //  6,291,456 B
    unsigned short* WoT     = (unsigned short*)(ws +  6291456);  //  2,097,152 B
    float*          biasP   = (float*)         (ws +  8388608);  //     12,288 B
    float*          scores  = (float*)         (ws +  8400896);  //  1,048,576 B
    unsigned*       VxP     = (unsigned*)      (ws +  9449472);  // 33,554,432 B
    unsigned*       concatP = (unsigned*)      (ws + 43003904);  // 33,554,432 B
    unsigned short* xbf     = (unsigned short*)(ws + 76558336);  // 33,554,432 B
    // total workspace used: 110,112,768 bytes

    pack_x<<<8192, 256, 0, stream>>>(x, (uint4*)xbf);
    pack_qkv_w<<<16, 256, 0, stream>>>(Wk, Wq, Wv, bk, bq, bv, Wp, biasP);
    pack_wo<<<1024, 256, 0, stream>>>(Wo, WoT);
    qkv_score_v<<<dim3(128, 16), 256, 98304, stream>>>(xbf, Wp, biasP, scores, VxP);
    softmax_scale<<<64, 256, 0, stream>>>(scores, VxP, concatP);
    out_gemm<<<dim3(128, 8), 256, 32768, stream>>>(
        (const unsigned short*)concatP, WoT, bo, out);
}
